// TGNUniMP_48670569398896
// MI455X (gfx1250) — matrix operand order repaired and checked by reading, not compiled, not run on hardware
//
#include <hip/hip_runtime.h>
#include <hip/hip_bf16.h>
#include <stdint.h>

// ---------------------------------------------------------------------------
// Types for WMMA fragments (gfx1250, wave32)
// ---------------------------------------------------------------------------
typedef __attribute__((ext_vector_type(16))) __bf16   v16bf;
typedef __attribute__((ext_vector_type(8)))  float    v8f;
typedef __attribute__((ext_vector_type(4)))  uint32_t u32x4;

union FragBF { u32x4 q[2]; v16bf v; };

__device__ __forceinline__ uint16_t f2bf(float f) {
    uint32_t u = __float_as_uint(f);
    u += 0x7FFFu + ((u >> 16) & 1u);     // round-to-nearest-even
    return (uint16_t)(u >> 16);
}
__device__ __forceinline__ float bf2f(uint16_t h) {
    return __uint_as_float(((uint32_t)h) << 16);
}

// A-fragment from LDS: 16x32 bf16 tile at k0 within a staged [16, ldk] tile.
// ISA 7.12.2: lane l holds row m=l%16; halves 0..7 = K=(l/16)*8..+7,
// halves 8..15 = K=16+(l/16)*8..+7.
__device__ __forceinline__ v16bf load_a_frag_lds(const uint16_t* As,
                                                 int ldk, int k0, int lane) {
    const int m = lane & 15, g = lane >> 4;
    const uint16_t* p = As + m * ldk + k0 + g * 8;
    FragBF f;
    f.q[0] = *(const u32x4*)p;
    f.q[1] = *(const u32x4*)(p + 16);
    return f.v;
}

// B-fragment: 32x16 bf16 where B[k][n] = W[n][k] (W row-major [Nout, K]).
// Lane l holds column n=l%16, halves 0..7 = K=(l/16)*8..+7, halves 8..15 = K=16+(l/16)*8..+7.
__device__ __forceinline__ v16bf load_b_frag(const uint16_t* __restrict__ W,
                                             int ldw, int col0, int k0, int lane) {
    const int n = lane & 15, g = lane >> 4;
    const uint16_t* p = W + (size_t)(col0 + n) * ldw + k0 + g * 8;
    FragBF f;
    f.q[0] = *(const u32x4*)p;
    f.q[1] = *(const u32x4*)(p + 16);
    return f.v;
}

// Async copy of one 16B chunk global -> LDS (per-lane), tracked by ASYNCcnt.
__device__ __forceinline__ void async_copy16(uint32_t lds_addr, const void* gptr) {
    uint64_t ga = (uint64_t)(uintptr_t)gptr;
    asm volatile("global_load_async_to_lds_b128 %0, %1, off"
                 :: "v"(lds_addr), "v"(ga) : "memory");
}
__device__ __forceinline__ void async_wait0() {
    asm volatile("s_wait_asynccnt 0x0" ::: "memory");
}

// ---------------------------------------------------------------------------
// Elementwise prep kernels
// ---------------------------------------------------------------------------
__global__ void k_f32_to_bf16(const float* __restrict__ s,
                              uint16_t* __restrict__ d, int n) {
    int i = blockIdx.x * blockDim.x + threadIdx.x;
    if (i < n) d[i] = f2bf(s[i]);
}

// e_attr[e][j] = j<64 ? cos(t[e]*w[j]+b[j]) : edge_feats[e][j-64], bf16 out
__global__ void k_edge_attr(const float* __restrict__ t,
                            const float* __restrict__ ef,
                            const float* __restrict__ wt,
                            const float* __restrict__ bt,
                            uint16_t* __restrict__ ea, int total) {
    int i = blockIdx.x * blockDim.x + threadIdx.x;
    if (i >= total) return;
    int e = i / 192, j = i - e * 192;
    float v = (j < 64) ? __cosf(t[e] * wt[j] + bt[j])
                       : ef[(size_t)e * 128 + (j - 64)];
    ea[i] = f2bf(v);
}

__global__ void k_init_softmax(float* __restrict__ m, float* __restrict__ den, int n) {
    int i = blockIdx.x * blockDim.x + threadIdx.x;
    if (i < n) { m[i] = -INFINITY; den[i] = 0.0f; }
}

// ---------------------------------------------------------------------------
// WMMA GEMM: node projections.  A=[Nn,256] bf16, W=[1024,256] bf16 (Q|K|V|S),
// bias=[1024].  cols 0..767 -> qkv f32 [Nn,768]; cols 768..1023 -> d_out.
// Block = 8 waves; A tile (16x256, 8KB) staged once via async-to-LDS and
// shared by all waves; each wave computes 4 adjacent 16x16 N-tiles (16x64),
// reusing each A fragment across 4 WMMAs. grid = (Nn/16, 2).
// ---------------------------------------------------------------------------
__global__ __launch_bounds__(256) void k_gemm_nodes(
    const uint16_t* __restrict__ A, const uint16_t* __restrict__ W,
    const float* __restrict__ bias,
    float* __restrict__ qkv, float* __restrict__ dout) {
    __shared__ uint16_t As[16 * 256];            // 8 KB
    const int tid  = threadIdx.x;
    const int lane = tid & 31;
    const int wave = tid >> 5;
    const int row0 = blockIdx.x * 16;

    // stage A tile: 512 x 16B chunks, 2 per thread, async global->LDS
    {
        const uint16_t* g = A + (size_t)row0 * 256;
        async_copy16((uint32_t)(uintptr_t)&As[tid * 8],         g + tid * 8);
        async_copy16((uint32_t)(uintptr_t)&As[(tid + 256) * 8], g + (tid + 256) * 8);
        async_wait0();
    }
    __syncthreads();

    const int col0 = (blockIdx.y * 32 + wave * 4) * 16;   // 4 tiles: col0..col0+63
    v8f acc[4] = {};
#pragma unroll
    for (int k0 = 0; k0 < 256; k0 += 32) {
        v16bf a = load_a_frag_lds(As, 256, k0, lane);
#pragma unroll
        for (int j = 0; j < 4; ++j) {
            v16bf b = load_b_frag(W, 256, col0 + j * 16, k0, lane);
            acc[j] = __builtin_amdgcn_wmma_f32_16x16x32_bf16(
                false, a, false, b, (short)0, acc[j], false, false);
        }
    }
    // C/D layout: lane l -> col = l%16, rows = (l/16)*8 + r
    const int rbase = row0 + (lane >> 4) * 8;
#pragma unroll
    for (int j = 0; j < 4; ++j) {
        const int col  = col0 + j * 16 + (lane & 15);
        const float bv = bias[col];
        if (col < 768) {
#pragma unroll
            for (int r = 0; r < 8; ++r)
                qkv[(size_t)(rbase + r) * 768 + col] = acc[j][r] + bv;
        } else {
#pragma unroll
            for (int r = 0; r < 8; ++r)
                dout[(size_t)(rbase + r) * 256 + (col - 768)] = acc[j][r] + bv;
        }
    }
}

// WMMA GEMM: edge projection. A=e_attr [E,192] bf16, W=We [256,192] bf16,
// out ep bf16 [E,256]. A tile (16x192, 6KB) staged via async-to-LDS;
// each of 8 waves computes 2 N-tiles (16 tiles total). grid = (E/16, 1).
__global__ __launch_bounds__(256) void k_gemm_edges(
    const uint16_t* __restrict__ A, const uint16_t* __restrict__ W,
    uint16_t* __restrict__ ep) {
    __shared__ uint16_t As[16 * 192];            // 6 KB
    const int tid  = threadIdx.x;
    const int lane = tid & 31;
    const int wave = tid >> 5;
    const int row0 = blockIdx.x * 16;

    // stage A tile: 384 x 16B chunks
    {
        const uint16_t* g = A + (size_t)row0 * 192;
        async_copy16((uint32_t)(uintptr_t)&As[tid * 8], g + tid * 8);
        if (tid < 128)
            async_copy16((uint32_t)(uintptr_t)&As[(tid + 256) * 8], g + (tid + 256) * 8);
        async_wait0();
    }
    __syncthreads();

    const int col0 = wave * 32;                  // 2 tiles per wave
    v8f acc[2] = {};
#pragma unroll
    for (int k0 = 0; k0 < 192; k0 += 32) {
        v16bf a = load_a_frag_lds(As, 192, k0, lane);
#pragma unroll
        for (int j = 0; j < 2; ++j) {
            v16bf b = load_b_frag(W, 192, col0 + j * 16, k0, lane);
            acc[j] = __builtin_amdgcn_wmma_f32_16x16x32_bf16(
                false, a, false, b, (short)0, acc[j], false, false);
        }
    }
    const int rbase = row0 + (lane >> 4) * 8;
#pragma unroll
    for (int j = 0; j < 2; ++j) {
        const int col = col0 + j * 16 + (lane & 15);
#pragma unroll
        for (int r = 0; r < 8; ++r)
            ep[(size_t)(rbase + r) * 256 + col] = f2bf(acc[j][r]);
    }
}

// ---------------------------------------------------------------------------
// Attention / segment softmax / scatter
// ---------------------------------------------------------------------------
__device__ __forceinline__ void atomicMaxF(float* addr, float val) {
    if (val >= 0.0f) atomicMax((int*)addr, __float_as_int(val));
    else             atomicMin((unsigned int*)addr, (unsigned int)__float_as_int(val));
}

// one thread per (edge, head): alpha = <q[dst], k[src]+ep> / sqrt(32); seg-max
__global__ void k_alpha(const int* __restrict__ src, const int* __restrict__ dst,
                        const float* __restrict__ qkv, const uint16_t* __restrict__ ep,
                        float* __restrict__ alpha, float* __restrict__ mbuf, int total) {
    int t = blockIdx.x * blockDim.x + threadIdx.x;
    if (t >= total) return;
    int e = t >> 3, h = t & 7;
    int s = src[e], d = dst[e];
    const float*    qp = qkv + (size_t)d * 768 + h * 32;          // Q
    const float*    kp = qkv + (size_t)s * 768 + 256 + h * 32;    // K
    const uint16_t* pp = ep  + (size_t)e * 256 + h * 32;
    float acc = 0.0f;
#pragma unroll
    for (int c = 0; c < 32; ++c) acc += qp[c] * (kp[c] + bf2f(pp[c]));
    acc *= 0.17677669529663689f;  // 1/sqrt(32)
    alpha[t] = acc;
    atomicMaxF(&mbuf[(size_t)d * 8 + h], acc);
}

__global__ void k_expsum(const int* __restrict__ dst,
                         float* __restrict__ alpha, const float* __restrict__ mbuf,
                         float* __restrict__ den, int total) {
    int t = blockIdx.x * blockDim.x + threadIdx.x;
    if (t >= total) return;
    int e = t >> 3, h = t & 7;
    int d = dst[e];
    float a = __expf(alpha[t] - mbuf[(size_t)d * 8 + h]);
    alpha[t] = a;
    atomicAdd(&den[(size_t)d * 8 + h], a);
}

// one thread per (edge, channel): out[dst] += a * (v[src] + ep)
__global__ void k_scatter(const int* __restrict__ src, const int* __restrict__ dst,
                          const float* __restrict__ qkv, const uint16_t* __restrict__ ep,
                          const float* __restrict__ alpha, const float* __restrict__ den,
                          float* __restrict__ dout, int total) {
    int t = blockIdx.x * blockDim.x + threadIdx.x;
    if (t >= total) return;
    int e = t >> 8, c = t & 255, h = c >> 5;
    int s = src[e], d = dst[e];
    float a = alpha[(size_t)e * 8 + h] / (den[(size_t)d * 8 + h] + 1e-16f);
    float v = qkv[(size_t)s * 768 + 512 + c] + bf2f(ep[(size_t)e * 256 + c]);
    atomicAdd(&dout[(size_t)d * 256 + c], a * v);
}

// ---------------------------------------------------------------------------
// Host-side launch
// ---------------------------------------------------------------------------
extern "C" void kernel_launch(void* const* d_in, const int* in_sizes, int n_in,
                              void* d_out, int out_size, void* d_ws, size_t ws_size,
                              hipStream_t stream) {
    const int E  = in_sizes[0] / 2;        // 320000
    const int Nn = in_sizes[3] / 256;      // 50000

    const int*   et    = (const int*)d_in[0];
    const int*   src   = et;
    const int*   dst   = et + E;
    const float* efeat = (const float*)d_in[1];
    const float* times = (const float*)d_in[2];
    const float* nodef = (const float*)d_in[3];
    const float* wt    = (const float*)d_in[4];
    const float* bt    = (const float*)d_in[5];
    const float* Wq = (const float*)d_in[6];  const float* bq = (const float*)d_in[7];
    const float* Wk = (const float*)d_in[8];  const float* bk = (const float*)d_in[9];
    const float* Wv = (const float*)d_in[10]; const float* bv = (const float*)d_in[11];
    const float* We = (const float*)d_in[12];
    const float* Wsk = (const float*)d_in[13]; const float* bs = (const float*)d_in[14];
    float* dout = (float*)d_out;

    // ---- workspace layout ----
    char* ws = (char*)d_ws;
    size_t off = 0;
    uint16_t* node_bf = (uint16_t*)(ws + off); off += (size_t)Nn * 256 * 2;   // 25.6 MB
    uint16_t* Wn_bf   = (uint16_t*)(ws + off); off += (size_t)1024 * 256 * 2; // 0.5 MB
    uint16_t* We_bf   = (uint16_t*)(ws + off); off += (size_t)256 * 192 * 2;  // 96 KB
    float*    bias_n  = (float*)   (ws + off); off += (size_t)1024 * 4;
    uint16_t* eattr   = (uint16_t*)(ws + off); off += (size_t)E * 192 * 2;    // 123 MB
    float*    qkv     = (float*)   (ws + off); off += (size_t)Nn * 768 * 4;   // 154 MB
    uint16_t* ep      = (uint16_t*)(ws + off); off += (size_t)E * 256 * 2;    // 164 MB
    float*    alpha   = (float*)   (ws + off); off += (size_t)E * 8 * 4;      // 10 MB
    float*    mbuf    = (float*)   (ws + off); off += (size_t)Nn * 8 * 4;
    float*    den     = (float*)   (ws + off); off += (size_t)Nn * 8 * 4;

    const int B = 256;
    // 1. bf16 conversions
    {   int n = Nn * 256;
        k_f32_to_bf16<<<(n + B - 1) / B, B, 0, stream>>>(nodef, node_bf, n); }
    {   int n = 256 * 256;
        k_f32_to_bf16<<<(n + B - 1) / B, B, 0, stream>>>(Wq,  Wn_bf,         n);
        k_f32_to_bf16<<<(n + B - 1) / B, B, 0, stream>>>(Wk,  Wn_bf + 1 * n, n);
        k_f32_to_bf16<<<(n + B - 1) / B, B, 0, stream>>>(Wv,  Wn_bf + 2 * n, n);
        k_f32_to_bf16<<<(n + B - 1) / B, B, 0, stream>>>(Wsk, Wn_bf + 3 * n, n); }
    {   int n = 256 * 192;
        k_f32_to_bf16<<<(n + B - 1) / B, B, 0, stream>>>(We, We_bf, n); }
    // bias concat [bq|bk|bv|bs]
    hipMemcpyAsync(bias_n + 0,   bq, 256 * sizeof(float), hipMemcpyDeviceToDevice, stream);
    hipMemcpyAsync(bias_n + 256, bk, 256 * sizeof(float), hipMemcpyDeviceToDevice, stream);
    hipMemcpyAsync(bias_n + 512, bv, 256 * sizeof(float), hipMemcpyDeviceToDevice, stream);
    hipMemcpyAsync(bias_n + 768, bs, 256 * sizeof(float), hipMemcpyDeviceToDevice, stream);

    // 2. time-encode + concat -> e_attr bf16
    {   int n = E * 192;
        k_edge_attr<<<(n + B - 1) / B, B, 0, stream>>>(times, efeat, wt, bt, eattr, n); }

    // 3. node GEMM (Q|K|V|skip) — skip+bias lands in d_out
    k_gemm_nodes<<<dim3(Nn / 16, 2), B, 0, stream>>>(node_bf, Wn_bf, bias_n, qkv, dout);

    // 4. edge GEMM -> ep bf16
    k_gemm_edges<<<dim3(E / 16, 1), B, 0, stream>>>(eattr, We_bf, ep);

    // 5. segment softmax + scatter
    {   int n = Nn * 8;
        k_init_softmax<<<(n + B - 1) / B, B, 0, stream>>>(mbuf, den, n); }
    {   int n = E * 8;
        k_alpha<<<(n + B - 1) / B, B, 0, stream>>>(src, dst, qkv, ep, alpha, mbuf, n);
        k_expsum<<<(n + B - 1) / B, B, 0, stream>>>(dst, alpha, mbuf, den, n); }
    {   int n = E * 256;
        k_scatter<<<(n + B - 1) / B, B, 0, stream>>>(src, dst, qkv, ep, alpha, den, dout, n); }
}
